// NLGF_40699110097034
// MI455X (gfx1250) — matrix operand order repaired and checked by reading, not compiled, not run on hardware
//
#include <hip/hip_runtime.h>
#include <hip/hip_bf16.h>
#include <math.h>

// ---------------------------------------------------------------------------
// CDNA5 (gfx1250) link-prediction GNN pipeline, bf16 WMMA everywhere.
//   prep_w : W, W2 (f32 row-major) -> Wt, W2t (bf16 column-major)
//   prep_x : X (f32) -> Xb (bf16 row-major), packed with v_perm_b32
//   k1     : XWt = (X @ W)^T  bf16            (wmma 16x16x32 bf16)
//   k2     : Z   = adj @ XW   bf16 row-major  (wmma, async-to-LDS B tile)
//   k3     : per-16-edge wave tile: h = relu([zi zj]) @ W2, logit=[h zmult]·W3
// ---------------------------------------------------------------------------

typedef __attribute__((ext_vector_type(16))) __bf16 v16bf;
typedef __attribute__((ext_vector_type(8)))  float  v8f;

union BFrag {
  v16bf v;
  uint4 q[2];
  unsigned u[8];
};

// one v_perm_b32: {lo.f32, hi.f32} -> packed bf16 pair (truncate)
__device__ __forceinline__ unsigned pk_bf16(unsigned hi, unsigned lo) {
  return __builtin_amdgcn_perm(hi, lo, 0x07060302u);
}
__device__ __forceinline__ unsigned short f2bf(float f) {
  unsigned u = __float_as_uint(f);
  u += 0x7fffu + ((u >> 16) & 1u);           // round-to-nearest-even
  return (unsigned short)(u >> 16);
}
__device__ __forceinline__ float bf2f(unsigned short h) {
  return __uint_as_float(((unsigned)h) << 16);
}
// packed bf16 relu in one VOP3P instruction
__device__ __forceinline__ unsigned relu_pk(unsigned u, unsigned z) {
  unsigned r;
  asm("v_pk_max_num_bf16 %0, %1, %2" : "=v"(r) : "v"(u), "v"(z));
  return r;
}
__device__ __forceinline__ void cvt8(BFrag& f, int off, uint4 a, uint4 b) {
  f.u[off + 0] = pk_bf16(a.y, a.x);
  f.u[off + 1] = pk_bf16(a.w, a.z);
  f.u[off + 2] = pk_bf16(b.y, b.x);
  f.u[off + 3] = pk_bf16(b.w, b.z);
}

// -------- prep: transpose W (512x256) and W2 (512x256) into bf16 col-major --
__global__ void prep_w(const float* __restrict__ W, const float* __restrict__ W2,
                       unsigned short* __restrict__ Wt, unsigned short* __restrict__ W2t) {
  int idx   = blockIdx.x * blockDim.x + threadIdx.x;   // 0..262143
  int which = idx >> 17;
  int r     = idx & 131071;
  int k     = r >> 8;                                  // 0..511
  int n     = r & 255;                                 // 0..255
  const float* src = which ? W2 : W;
  unsigned short* dst = which ? W2t : Wt;
  dst[n * 512 + k] = f2bf(src[k * 256 + n]);
}

// -------- prep: X f32 -> bf16 row-major, 8 elements/thread, vectorized ------
__global__ void prep_x(const float* __restrict__ X, unsigned short* __restrict__ Xb,
                       int total8) {
  int idx = blockIdx.x * blockDim.x + threadIdx.x;
  if (idx >= total8) return;
  const uint4* s = (const uint4*)(X + (size_t)idx * 8);
  uint4 a = s[0], b = s[1];
  uint4 o;
  o.x = pk_bf16(a.y, a.x);
  o.y = pk_bf16(a.w, a.z);
  o.z = pk_bf16(b.y, b.x);
  o.w = pk_bf16(b.w, b.z);
  ((uint4*)Xb)[idx] = o;
}

// -------- k1: XWt[col*Nn + row] = bf16( (X @ W)[row, col] ) -----------------
__global__ void k1_xw(const unsigned short* __restrict__ Xb,
                      const unsigned short* __restrict__ Wt,
                      unsigned short* __restrict__ XWt, int Nn, int wtiles) {
  int lane = threadIdx.x & 31;
  int wid  = blockIdx.x * (blockDim.x >> 5) + (threadIdx.x >> 5);
  if (wid >= wtiles) return;
  int mtile = wid >> 4;                  // 16 N-tiles per M-tile
  int ntile = wid & 15;
  int m0 = mtile * 16;
  int hl = (lane < 16) ? 0 : 1;
  int rowA = m0 + (lane & 15);
  int col  = ntile * 16 + (lane & 15);

  v8f acc;
#pragma unroll
  for (int r = 0; r < 8; ++r) acc[r] = 0.f;

  for (int t = 0; t < 16; ++t) {         // K = 512 in steps of 32
    int kb = t * 32;
    BFrag A, B;
    const unsigned short* ap = Xb + (size_t)rowA * 512 + kb + hl * 16;
    A.q[0] = ((const uint4*)ap)[0];
    A.q[1] = ((const uint4*)(ap + 8))[0];
    const uint4* bp = (const uint4*)(Wt + (size_t)col * 512 + kb + hl * 16);
    B.q[0] = bp[0]; B.q[1] = bp[1];
    acc = __builtin_amdgcn_wmma_f32_16x16x32_bf16(false, A.v, false, B.v,
                                                  (short)0, acc, false, false);
  }
#pragma unroll
  for (int r = 0; r < 8; ++r) {
    int row = m0 + r + hl * 8;
    XWt[(size_t)col * Nn + row] = f2bf(acc[r]);
  }
}

// -------- k2: Z[row*256+col] = bf16( (adj @ XW)[row, col] ) -----------------
// Block = 8 waves; each wave: 16 rows x 128 cols.
// B tile staged in LDS via GLOBAL_LOAD_ASYNC_TO_LDS_B128 (ASYNCcnt path).
__global__ void k2_z(const float* __restrict__ adj, const unsigned short* __restrict__ XWt,
                     unsigned short* __restrict__ Z, int Nn) {
  __shared__ __align__(16) unsigned short Bt[128 * 32];   // 128 cols x 32 K, 8 KB
  int lane = threadIdx.x & 31;
  int w    = threadIdx.x >> 5;
  int m0   = blockIdx.x * 128 + w * 16;
  int n0   = blockIdx.y * 128;
  int hl   = (lane < 16) ? 0 : 1;
  int rowA = m0 + (lane & 15);
  if (rowA >= Nn) rowA = Nn - 1;         // clamp; stores guarded below

  v8f acc[8];
#pragma unroll
  for (int nt = 0; nt < 8; ++nt)
#pragma unroll
    for (int r = 0; r < 8; ++r) acc[nt][r] = 0.f;

  int cfill = threadIdx.x >> 1;          // 0..127
  int half  = threadIdx.x & 1;
  unsigned ldsoff =
      (unsigned)(reinterpret_cast<uintptr_t>(&Bt[cfill * 32 + half * 16]));
  int nsteps = (Nn + 31) / 32;
  const uint4 z4 = make_uint4(0, 0, 0, 0);

  for (int s = 0; s < nsteps; ++s) {
    int kb = s * 32;
    __syncthreads();                     // previous tile fully consumed
    {  // fill B tile: XWt[(n0+c)*Nn + kb .. kb+32) -> LDS
      int g = kb + half * 16;
      const unsigned short* src = XWt + (size_t)(n0 + cfill) * Nn + g;
      if (g + 16 <= Nn) {
        // async memory->LDS; INST_OFFSET applies to both LDS and global addr
        asm volatile(
            "global_load_async_to_lds_b128 %0, %1, off\n\t"
            "global_load_async_to_lds_b128 %0, %1, off offset:16"
            :: "v"(ldsoff), "v"(src) : "memory");
      } else {                           // K remainder: zero-padded sync fill
        uint4* dst = (uint4*)&Bt[cfill * 32 + half * 16];
        const uint4* sp = (const uint4*)src;
        dst[0] = (g + 8  <= Nn) ? sp[0] : z4;
        dst[1] = (g + 16 <= Nn) ? sp[1] : z4;
      }
      asm volatile("s_wait_asynccnt 0" ::: "memory");
    }
    __syncthreads();                     // tile visible to all waves

    BFrag A;
    {  // A fragment from adj (f32 -> bf16 via v_perm), K-remainder zero-padded
      const float* ap = adj + (size_t)rowA * Nn + kb + hl * 16;
      int c0 = kb + hl * 16;
      int c1 = c0 + 8;
      uint4 a0 = (c0 + 8 <= Nn) ? ((const uint4*)ap)[0] : z4;
      uint4 a1 = (c0 + 8 <= Nn) ? ((const uint4*)ap)[1] : z4;
      uint4 b0 = (c1 + 8 <= Nn) ? ((const uint4*)(ap + 8))[0] : z4;
      uint4 b1 = (c1 + 8 <= Nn) ? ((const uint4*)(ap + 8))[1] : z4;
      cvt8(A, 0, a0, a1);
      cvt8(A, 4, b0, b1);
    }
    if (s + 1 < nsteps)
      __builtin_prefetch(adj + (size_t)rowA * Nn + kb + 32, 0, 1);

#pragma unroll
    for (int nt = 0; nt < 8; ++nt) {
      BFrag B;
      const uint4* bp = (const uint4*)&Bt[(nt * 16 + (lane & 15)) * 32 + hl * 16];
      B.q[0] = bp[0]; B.q[1] = bp[1];
      acc[nt] = __builtin_amdgcn_wmma_f32_16x16x32_bf16(false, A.v, false, B.v,
                                                        (short)0, acc[nt], false, false);
    }
  }

#pragma unroll
  for (int nt = 0; nt < 8; ++nt) {
    int col = n0 + nt * 16 + (lane & 15);
#pragma unroll
    for (int r = 0; r < 8; ++r) {
      int row = m0 + r + hl * 8;
      if (row < Nn) Z[(size_t)row * 256 + col] = f2bf(acc[nt][r]);
    }
  }
}

// -------- k3: per wave, 16 edges: h = relu([zi zj]) @ W2; logit; sigmoid ----
__global__ void k3_edges(const int* __restrict__ te, const int* __restrict__ tf,
                         const unsigned short* __restrict__ Zb,
                         const unsigned short* __restrict__ W2t,
                         const float* __restrict__ W3,
                         float* __restrict__ out, int E, int TE) {
  int lane = threadIdx.x & 31;
  int wid  = blockIdx.x * (blockDim.x >> 5) + (threadIdx.x >> 5);
  int e0   = wid * 16;
  if (e0 >= TE) return;
  int hl   = (lane < 16) ? 0 : 1;
  int eidx = e0 + (lane & 15);
  int el   = (eidx < TE) ? eidx : (TE - 1);
  int vi, vj;
  if (el < E) { vi = te[2 * el]; vj = te[2 * el + 1]; }
  else        { int r = el - E; vi = tf[2 * r]; vj = tf[2 * r + 1]; }

  unsigned zero = 0u;                     // VGPR 0 for v_pk_max_num_bf16
  v8f acc[16];                            // 16 N-tiles (N=256), 128 VGPRs
#pragma unroll
  for (int n = 0; n < 16; ++n)
#pragma unroll
    for (int r = 0; r < 8; ++r) acc[n][r] = 0.f;

  for (int t = 0; t < 16; ++t) {          // K = 512 (zi: t<8, zj: t>=8)
    int sel  = (t < 8) ? vi : vj;
    int base = (t & 7) * 32 + hl * 16;
    BFrag A;
    const uint4* p0 = (const uint4*)(Zb + (size_t)sel * 256 + base);
    const uint4* p1 = (const uint4*)(Zb + (size_t)sel * 256 + base + 8);
    A.q[0] = p0[0]; A.q[1] = p1[0];
#pragma unroll
    for (int k = 0; k < 8; ++k) A.u[k] = relu_pk(A.u[k], zero);  // relu(z_pair)
    int koff = t * 32 + hl * 16;
#pragma unroll
    for (int n = 0; n < 16; ++n) {
      BFrag B;
      const uint4* bp = (const uint4*)(W2t + (size_t)(n * 16 + (lane & 15)) * 512 + koff);
      B.q[0] = bp[0]; B.q[1] = bp[1];
      acc[n] = __builtin_amdgcn_wmma_f32_16x16x32_bf16(false, A.v, false, B.v,
                                                       (short)0, acc[n], false, false);
    }
  }

  // logit += sum_n h[m, n] * W3[n] ; lane owns its column set
  float w3c[16];
#pragma unroll
  for (int n = 0; n < 16; ++n) w3c[n] = W3[n * 16 + (lane & 15)];
  float lg[8];
#pragma unroll
  for (int r = 0; r < 8; ++r) {
    float s = 0.f;
#pragma unroll
    for (int n = 0; n < 16; ++n) s += acc[n][r] * w3c[n];
    s += __shfl_xor(s, 1);
    s += __shfl_xor(s, 2);
    s += __shfl_xor(s, 4);
    s += __shfl_xor(s, 8);                // reduce within 16-lane half
    lg[r] = s;
  }

  // logit += sum_c zi[c]*zj[c]*W3[256+c]; then sigmoid
  float w3h[8];
#pragma unroll
  for (int q = 0; q < 8; ++q) w3h[q] = W3[256 + lane + q * 32];
  for (int m = 0; m < 16; ++m) {
    int im = __shfl(vi, m);
    int jm = __shfl(vj, m);
    float p = 0.f;
#pragma unroll
    for (int q = 0; q < 8; ++q) {
      int c = lane + q * 32;
      p += bf2f(Zb[(size_t)im * 256 + c]) * bf2f(Zb[(size_t)jm * 256 + c]) * w3h[q];
    }
    p += __shfl_xor(p, 1);
    p += __shfl_xor(p, 2);
    p += __shfl_xor(p, 4);
    p += __shfl_xor(p, 8);
    p += __shfl_xor(p, 16);
    bool owner = (m < 8) ? (lane == m) : (lane == m + 8);
    int eo = e0 + m;
    if (owner && eo < TE) {
      float logit = lg[m & 7] + p;
      out[eo] = 1.f / (1.f + expf(-logit));
    }
  }
}

// ---------------------------------------------------------------------------
extern "C" void kernel_launch(void* const* d_in, const int* in_sizes, int n_in,
                              void* d_out, int out_size, void* d_ws, size_t ws_size,
                              hipStream_t stream) {
  const float* X   = (const float*)d_in[0];
  const float* adj = (const float*)d_in[1];
  const int*   te  = (const int*)d_in[2];
  const int*   tf  = (const int*)d_in[3];
  const float* W   = (const float*)d_in[4];
  const float* W2  = (const float*)d_in[5];
  const float* W3  = (const float*)d_in[6];
  float* out = (float*)d_out;

  const int IN = 512, HID = 256;
  int Nn = in_sizes[0] / IN;            // 10000
  int E  = in_sizes[2] / 2;             // 400000
  int TE = E + in_sizes[3] / 2;         // 800000

  // workspace layout (256B-aligned chunks):
  //   Wt 256K | W2t 256K | Xb 2*IN*Nn | XWt 2*HID*Nn | Zb 2*HID*Nn
  char* ws = (char*)d_ws;
  size_t off = 0;
  unsigned short* Wt  = (unsigned short*)(ws + off); off += 262144;
  unsigned short* W2t = (unsigned short*)(ws + off); off += 262144;
  unsigned short* Xb  = (unsigned short*)(ws + off);
  off += (((size_t)IN * Nn * 2 + 255) & ~(size_t)255);
  unsigned short* XWt = (unsigned short*)(ws + off);
  off += (((size_t)HID * Nn * 2 + 255) & ~(size_t)255);
  unsigned short* Zb  = (unsigned short*)(ws + off);

  // 1) transpose+convert weights; convert X to bf16
  prep_w<<<1024, 256, 0, stream>>>(W, W2, Wt, W2t);
  int total8 = Nn * IN / 8;
  prep_x<<<(total8 + 255) / 256, 256, 0, stream>>>(X, Xb, total8);

  // 2) XWt = (X @ W)^T bf16 : (Nn/16)*16 wave tiles, 8 waves/block
  int wt1 = (Nn / 16) * 16;
  k1_xw<<<(wt1 + 7) / 8, 256, 0, stream>>>(Xb, Wt, XWt, Nn, wt1);

  // 3) Z = adj @ XW : blocks of 128 rows x 128 cols (HID=256 -> grid.y=2)
  dim3 g2((Nn + 127) / 128, (HID + 127) / 128);
  k2_z<<<g2, 256, 0, stream>>>(adj, XWt, Zb, Nn);

  // 4) edge MLP + logits + sigmoid : 16 edges per wave, 8 waves/block
  int wt3 = (TE + 15) / 16;
  k3_edges<<<(wt3 + 7) / 8, 256, 0, stream>>>(te, tf, Zb, W2t, W3, out, E, TE);
}
